// GraphContextGAT_88562225643677
// MI455X (gfx1250) — compile-verified, matrix operand order checked by reading, not run on hardware
//
#include <hip/hip_runtime.h>
#include <math.h>

typedef __attribute__((ext_vector_type(16))) __bf16 v16bf;
typedef __attribute__((ext_vector_type(8)))  float  v8f;
typedef __attribute__((ext_vector_type(4)))  float  v4f;
typedef __attribute__((ext_vector_type(4)))  unsigned int v4u;

#define NNODES 10000
#define NEDGES 80000
#define GFD    768
#define FDIM   512
#define NH     8
#define EF     (NEDGES + NNODES)
#define CDIV(a,b) (((a)+(b)-1)/(b))

__device__ __forceinline__ unsigned short f2bf(float f) {
    unsigned int u = __builtin_bit_cast(unsigned int, f);
    u += 0x7FFFu + ((u >> 16) & 1u);          // round-to-nearest-even
    return (unsigned short)(u >> 16);
}
__device__ __forceinline__ unsigned int pk2(float lo, float hi) {
    return (unsigned int)f2bf(lo) | ((unsigned int)f2bf(hi) << 16);
}
__device__ __forceinline__ unsigned int encf(float f) {
    unsigned int u = __builtin_bit_cast(unsigned int, f);
    return (u & 0x80000000u) ? ~u : (u | 0x80000000u);
}
__device__ __forceinline__ float decf(unsigned int u) {
    unsigned int b = (u & 0x80000000u) ? (u & 0x7FFFFFFFu) : ~u;
    return __builtin_bit_cast(float, b);
}
__device__ __forceinline__ void edge_sd(const int* __restrict__ edges, int e, int& s, int& d) {
    if (e < NEDGES) { s = edges[2*e]; d = edges[2*e+1]; }
    else            { s = d = e - NEDGES; }
}
__device__ __forceinline__ v4f lk4(v4f a) {
    v4f r;
    #pragma unroll
    for (int i = 0; i < 4; ++i) r[i] = (a[i] >= 0.f) ? a[i] : 0.01f * a[i];
    return r;
}

// ---------------- bf16 WMMA GEMM: C = op(A[gidx]) * B (+bias) ----------------
// Block tile: BM=128, BN=64, BK=32. 256 threads = 8 waves (4x2 grid),
// each wave computes a 32x32 output = 2x2 v_wmma_f32_16x16x32_bf16 tiles.
#define BM 128
#define BN 64
#define LDSP 20   // padded row stride (u32): keeps 16B alignment, conflict-free

__global__ __launch_bounds__(256)
void k_wmma_gemm(const float* __restrict__ A, const int* __restrict__ gidx, int lda,
                 const float* __restrict__ B, int ldb,
                 const float* __restrict__ bias,
                 float* __restrict__ C, int ldc,
                 int M, int N, int K, int leakyA)
{
    __shared__ unsigned int As[BM][LDSP];  // [m][kpair] packed bf16 (k even lo, odd hi)
    __shared__ unsigned int Bs[BN][LDSP];  // [n][kpair] packed bf16

    const int tid     = threadIdx.x;
    const int rowBase = blockIdx.y * BM;
    const int colBase = blockIdx.x * BN;

    // A staging: thread -> (row = tid>>1, k-half = tid&1 -> 16 floats)
    const int arow = tid >> 1;
    const int ah   = tid & 1;
    const int agm  = rowBase + arow;
    const bool avalid = agm < M;
    const int  ridx = avalid ? (gidx ? gidx[agm] : agm) : 0;
    const float* aptr = A + (size_t)ridx * lda + ah * 16;

    // B staging: thread -> (kpair = tid>>4, 4 cols at (tid&15)*4)
    const int bkp = tid >> 4;
    const int bn0 = (tid & 15) * 4;
    const float* bptr = B + (size_t)(2 * bkp) * ldb + colBase + bn0;

    const int lane = tid & 31, wave = tid >> 5;
    const int wm = wave >> 1, wn = wave & 1;     // 4x2 wave grid
    const int l15 = lane & 15, lhi = lane >> 4;

    v8f acc00 = {}, acc01 = {}, acc10 = {}, acc11 = {};

    for (int k0 = 0; k0 < K; k0 += 32) {
        // ---- stage A (128x32 floats -> packed bf16 pairs)
        v4f z = {0.f, 0.f, 0.f, 0.f};
        v4f a0 = z, a1 = z, a2 = z, a3 = z;
        if (avalid) {
            const v4f* p = (const v4f*)(aptr + k0);
            a0 = p[0]; a1 = p[1]; a2 = p[2]; a3 = p[3];
            if (leakyA) { a0 = lk4(a0); a1 = lk4(a1); a2 = lk4(a2); a3 = lk4(a3); }
        }
        v4u w0 = { pk2(a0[0],a0[1]), pk2(a0[2],a0[3]), pk2(a1[0],a1[1]), pk2(a1[2],a1[3]) };
        v4u w1 = { pk2(a2[0],a2[1]), pk2(a2[2],a2[3]), pk2(a3[0],a3[1]), pk2(a3[2],a3[3]) };
        *(v4u*)&As[arow][ah * 8]     = w0;
        *(v4u*)&As[arow][ah * 8 + 4] = w1;

        // ---- stage B (32x64 floats -> transposed packed pairs [n][kp])
        const float* br = bptr + (size_t)k0 * ldb;
        v4f b0 = *(const v4f*)br;
        v4f b1 = *(const v4f*)(br + ldb);
        #pragma unroll
        for (int j = 0; j < 4; ++j) Bs[bn0 + j][bkp] = pk2(b0[j], b1[j]);

        __syncthreads();

        // ---- fragments: kpair order per ISA 16-bit A 16x32 layout
        // u[0..3] = kp lhi*4..+3 ; u[4..7] = kp 8+lhi*4..+3  (contiguous -> b128)
        union Fr { v16bf v; v4u q[2]; };
        Fr fa0, fa1, fb0, fb1;
        const int rA0 = wm * 32 + l15, rA1 = rA0 + 16;
        const int cB0 = wn * 32 + l15, cB1 = cB0 + 16;
        fa0.q[0] = *(const v4u*)&As[rA0][lhi * 4];
        fa0.q[1] = *(const v4u*)&As[rA0][8 + lhi * 4];
        fa1.q[0] = *(const v4u*)&As[rA1][lhi * 4];
        fa1.q[1] = *(const v4u*)&As[rA1][8 + lhi * 4];
        fb0.q[0] = *(const v4u*)&Bs[cB0][lhi * 4];
        fb0.q[1] = *(const v4u*)&Bs[cB0][8 + lhi * 4];
        fb1.q[0] = *(const v4u*)&Bs[cB1][lhi * 4];
        fb1.q[1] = *(const v4u*)&Bs[cB1][8 + lhi * 4];

        acc00 = __builtin_amdgcn_wmma_f32_16x16x32_bf16(false, fa0.v, false, fb0.v, (short)0, acc00, false, false);
        acc01 = __builtin_amdgcn_wmma_f32_16x16x32_bf16(false, fa0.v, false, fb1.v, (short)0, acc01, false, false);
        acc10 = __builtin_amdgcn_wmma_f32_16x16x32_bf16(false, fa1.v, false, fb0.v, (short)0, acc10, false, false);
        acc11 = __builtin_amdgcn_wmma_f32_16x16x32_bf16(false, fa1.v, false, fb1.v, (short)0, acc11, false, false);

        __syncthreads();
    }

    // ---- epilogue: C/D layout lanes0-15 -> M=r, lanes16-31 -> M=8+r, N=lane&15
    const int col0 = colBase + wn * 32 + l15;
    const int col1 = col0 + 16;
    const int mrow = rowBase + wm * 32 + (lhi << 3);
    const float bv0 = bias ? bias[col0] : 0.f;
    const float bv1 = bias ? bias[col1] : 0.f;
    #pragma unroll
    for (int r = 0; r < 8; ++r) {
        int m0 = mrow + r, m1 = m0 + 16;
        if (m0 < M) {
            C[(size_t)m0 * ldc + col0] = acc00[r] + bv0;
            C[(size_t)m0 * ldc + col1] = acc01[r] + bv1;
        }
        if (m1 < M) {
            C[(size_t)m1 * ldc + col0] = acc10[r] + bv0;
            C[(size_t)m1 * ldc + col1] = acc11[r] + bv1;
        }
    }
}

// ---------------- small kernels ----------------
__global__ void k_gather_rows(const float* __restrict__ emb, const int* __restrict__ idx,
                              float* __restrict__ dst, int rows, int width, int ldd)
{
    int t = blockIdx.x * blockDim.x + threadIdx.x;
    if (t >= rows * width) return;
    int n = t / width, k = t - n * width;
    dst[(size_t)n * ldd + k] = emb[(size_t)idx[n] * width + k];
}

__global__ void k_deg(const int* __restrict__ edges, float* __restrict__ deg)
{
    int e = blockIdx.x * blockDim.x + threadIdx.x;
    if (e < NEDGES) atomicAdd(&deg[edges[2*e+1]], 1.0f);
}

__global__ __launch_bounds__(256)
void k_loop_sum(const int* __restrict__ edges, float* __restrict__ ea)
{
    int e = blockIdx.x;
    int d = edges[2*e+1];
    const float* sr = ea + (size_t)e * GFD;
    float* dr = ea + (size_t)(NEDGES + d) * GFD;
    for (int k = threadIdx.x; k < GFD; k += 256)
        atomicAdd(&dr[k], sr[k]);
}

__global__ void k_loop_div(float* __restrict__ ea, const float* __restrict__ deg)
{
    int t = blockIdx.x * blockDim.x + threadIdx.x;
    if (t >= NNODES * GFD) return;
    int n = t / GFD;
    float dg = deg[n]; dg = dg > 1.f ? dg : 1.f;
    ea[(size_t)NEDGES * GFD + t] /= dg;
}

// ve[k*8+h] = sum_j W_e[k][h*c+j] * a_e[h][j]
__global__ void k_fold_ve(const float* __restrict__ We, const float* __restrict__ ae,
                          float* __restrict__ ve, int c)
{
    int t = blockIdx.x * blockDim.x + threadIdx.x;
    if (t >= GFD * NH) return;
    int k = t >> 3, hh = t & 7;
    const float* w = We + (size_t)k * (NH * c) + hh * c;
    const float* a = ae + hh * c;
    float s = 0.f;
    for (int j = 0; j < c; ++j) s += w[j] * a[j];
    ve[t] = s;
}

__global__ void k_head_terms(const float* __restrict__ h, const float* __restrict__ asrc,
                             const float* __restrict__ adst, float* __restrict__ st,
                             float* __restrict__ dt, int c)
{
    int t = blockIdx.x * blockDim.x + threadIdx.x;
    if (t >= NNODES * NH) return;
    int n = t >> 3, hh = t & 7;
    const float* hr = h + (size_t)n * (NH * c) + hh * c;
    float s = 0.f, d = 0.f;
    for (int j = 0; j < c; ++j) { float v = hr[j]; s += v * asrc[hh*c+j]; d += v * adst[hh*c+j]; }
    st[t] = s; dt[t] = d;
}

__global__ void k_ae_term(const float* __restrict__ ea, const float* __restrict__ ve,
                          float* __restrict__ out)
{
    int t = blockIdx.x * blockDim.x + threadIdx.x;
    if (t >= EF * NH) return;
    int e = t >> 3, hh = t & 7;
    const float* row = ea + (size_t)e * GFD;
    float s = 0.f;
    for (int k = 0; k < GFD; ++k) s += row[k] * ve[k*8 + hh];
    out[t] = s;
}

__global__ void k_logits_max(const int* __restrict__ edges, const float* __restrict__ st,
                             const float* __restrict__ dt, const float* __restrict__ aet,
                             float* __restrict__ lg, unsigned int* __restrict__ amax)
{
    int t = blockIdx.x * blockDim.x + threadIdx.x;
    if (t >= EF * NH) return;
    int e = t >> 3, hh = t & 7, s, d;
    edge_sd(edges, e, s, d);
    float a = st[s*8+hh] + dt[d*8+hh] + aet[t];
    a = (a >= 0.f) ? a : 0.2f * a;
    lg[t] = a;
    atomicMax(&amax[d*8+hh], encf(a));
}

__global__ void k_exp_den(const int* __restrict__ edges, float* __restrict__ lg,
                          const unsigned int* __restrict__ amax, float* __restrict__ den)
{
    int t = blockIdx.x * blockDim.x + threadIdx.x;
    if (t >= EF * NH) return;
    int e = t >> 3, hh = t & 7, s, d;
    edge_sd(edges, e, s, d);
    float ea = __expf(lg[t] - decf(amax[d*8+hh]));
    lg[t] = ea;
    atomicAdd(&den[d*8+hh], ea);
}

__global__ __launch_bounds__(256)
void k_aggregate(const int* __restrict__ edges, const float* __restrict__ h,
                 const float* __restrict__ lg, const float* __restrict__ den,
                 float* __restrict__ agg, int c, int cShift)
{
    int e = blockIdx.x;
    int s, d; edge_sd(edges, e, s, d);
    __shared__ float al[NH];
    if (threadIdx.x < NH)
        al[threadIdx.x] = lg[(size_t)e*NH + threadIdx.x] /
                          (den[(size_t)d*NH + threadIdx.x] + 1e-16f);
    __syncthreads();
    int Hc = NH * c;
    const float* hs = h + (size_t)s * Hc;
    float* ad = agg + (size_t)d * Hc;
    for (int i = threadIdx.x; i < Hc; i += 256)
        atomicAdd(&ad[i], hs[i] * al[i >> cShift]);
}

__global__ void k_fin_concat(const float* __restrict__ agg, const float* __restrict__ b,
                             float* __restrict__ xo, int Hc)
{
    int t = blockIdx.x * blockDim.x + threadIdx.x;
    if (t >= NNODES * Hc) return;
    float v = agg[t] + b[t % Hc];
    xo[t] = v > 0.f ? v : 0.f;
}

__global__ void k_fin_mean(const float* __restrict__ agg, const float* __restrict__ b,
                           float* __restrict__ xo)
{
    int t = blockIdx.x * blockDim.x + threadIdx.x;
    if (t >= NNODES * FDIM) return;
    int n = t / FDIM, j = t - n * FDIM;
    float s = 0.f;
    for (int hh = 0; hh < NH; ++hh) s += agg[(size_t)n*NH*FDIM + hh*FDIM + j];
    xo[t] = s * (1.f / NH) + b[j];
}

__global__ void k_scatter_out(const float* __restrict__ img, const float* __restrict__ x3,
                              const int* __restrict__ bn, float* __restrict__ out)
{
    int t = blockIdx.x * blockDim.x + threadIdx.x;
    if (t >= 16 * 177 * 512) return;
    int f = t & 511; int r = t >> 9; int tt = r % 177; int b = r / 177;
    float v;
    if (tt < 49) v = img[((size_t)tt * 16 + b) * 512 + f];
    else         v = x3[(size_t)bn[b * 128 + (tt - 49)] * 512 + f];
    out[t] = v;
}

// ---------------- host ----------------
extern "C" void kernel_launch(void* const* d_in, const int* in_sizes, int n_in,
                              void* d_out, int out_size, void* d_ws, size_t ws_size,
                              hipStream_t stream)
{
    const int*   nodes      = (const int*)d_in[0];
    const int*   node_types = (const int*)d_in[1];
    const int*   edge_types = (const int*)d_in[2];
    const int*   edges      = (const int*)d_in[3];
    const int*   batched    = (const int*)d_in[4];
    const float* images     = (const float*)d_in[5];
    const float* emb        = (const float*)d_in[6];
    const float* W_img      = (const float*)d_in[7];
    const float* b_img      = (const float*)d_in[8];
    const float* W_down     = (const float*)d_in[9];
    const float* b_down     = (const float*)d_in[10];
    const float* W_eup      = (const float*)d_in[11];
    const float* b_eup      = (const float*)d_in[12];
    float* out = (float*)d_out;

    float* ws = (float*)d_ws;
    size_t o = 0;
    float* X0  = ws + o; o += (size_t)NNODES * 1024;
    float* X1  = ws + o; o += (size_t)NNODES * 1024;
    float* Hb  = ws + o; o += (size_t)NNODES * 4096;
    float* AG  = ws + o; o += (size_t)NNODES * 4096;
    float* EA  = ws + o; o += (size_t)EF * GFD;
    float* VE  = ws + o; o += (size_t)GFD * NH;
    float* ST  = ws + o; o += (size_t)NNODES * NH;
    float* DT  = ws + o; o += (size_t)NNODES * NH;
    float* AE  = ws + o; o += (size_t)EF * NH;
    float* LG  = ws + o; o += (size_t)EF * NH;
    unsigned int* AMAX = (unsigned int*)(ws + o); o += (size_t)NNODES * NH;
    float* DEN = ws + o; o += (size_t)NNODES * NH;
    float* DEG = ws + o; o += (size_t)NNODES;
    float* IMG = ws + o; o += (size_t)49 * 16 * 512;

    // ---- node features x0 = [emb[nodes], emb[node_types] @ W_down + b_down]  (ld = 768)
    k_gather_rows<<<CDIV(NNODES*512,256),256,0,stream>>>(emb, nodes, X0, NNODES, 512, GFD);
    k_wmma_gemm<<<dim3(256/BN, CDIV(NNODES,BM)),256,0,stream>>>(
        emb, node_types, 512, W_down, 256, b_down, X0 + 512, GFD, NNODES, 256, 512, 0);

    // ---- eattr = emb[edge_types] @ W_eup + b_eup  -> EA rows [0, NEDGES)
    k_wmma_gemm<<<dim3(GFD/BN, CDIV(NEDGES,BM)),256,0,stream>>>(
        emb, edge_types, 512, W_eup, GFD, b_eup, EA, GFD, NEDGES, GFD, 512, 0);

    // ---- degree + self-loop attrs -> EA rows [NEDGES, NEDGES+NNODES)
    hipMemsetAsync(DEG, 0, (size_t)NNODES * 4, stream);
    k_deg<<<CDIV(NEDGES,256),256,0,stream>>>(edges, DEG);
    hipMemsetAsync(EA + (size_t)NEDGES * GFD, 0, (size_t)NNODES * GFD * 4, stream);
    k_loop_sum<<<NEDGES,256,0,stream>>>(edges, EA);
    k_loop_div<<<CDIV(NNODES*GFD,256),256,0,stream>>>(EA, DEG);

    // ---- GAT layers
    const int din[3]  = {GFD, 1024, 1024};
    const int cdim[3] = {128, 128, 512};
    float* xin = X0;
    for (int l = 0; l < 3; ++l) {
        const float* W     = (const float*)d_in[13 + 6*l + 0];
        const float* a_src = (const float*)d_in[13 + 6*l + 1];
        const float* a_dst = (const float*)d_in[13 + 6*l + 2];
        const float* We    = (const float*)d_in[13 + 6*l + 3];
        const float* ae    = (const float*)d_in[13 + 6*l + 4];
        const float* bvec  = (const float*)d_in[13 + 6*l + 5];
        int c = cdim[l], Hc = NH * c, K = din[l];
        int cShift = (c == 128) ? 7 : 9;

        k_wmma_gemm<<<dim3(Hc/BN, CDIV(NNODES,BM)),256,0,stream>>>(
            xin, nullptr, K, W, Hc, nullptr, Hb, Hc, NNODES, Hc, K, 0);
        k_head_terms<<<CDIV(NNODES*NH,256),256,0,stream>>>(Hb, a_src, a_dst, ST, DT, c);
        k_fold_ve<<<CDIV(GFD*NH,256),256,0,stream>>>(We, ae, VE, c);
        k_ae_term<<<CDIV(EF*NH,256),256,0,stream>>>(EA, VE, AE);
        hipMemsetAsync(AMAX, 0, (size_t)NNODES * NH * 4, stream);
        hipMemsetAsync(DEN,  0, (size_t)NNODES * NH * 4, stream);
        k_logits_max<<<CDIV(EF*NH,256),256,0,stream>>>(edges, ST, DT, AE, LG, AMAX);
        k_exp_den<<<CDIV(EF*NH,256),256,0,stream>>>(edges, LG, AMAX, DEN);
        hipMemsetAsync(AG, 0, (size_t)NNODES * Hc * 4, stream);
        k_aggregate<<<EF,256,0,stream>>>(edges, Hb, LG, DEN, AG, c, cShift);
        if (l < 2) {
            float* xo = (l == 0) ? X1 : X0;
            k_fin_concat<<<CDIV(NNODES*Hc,256),256,0,stream>>>(AG, bvec, xo, Hc);
            xin = xo;
        } else {
            k_fin_mean<<<CDIV(NNODES*FDIM,256),256,0,stream>>>(AG, bvec, X1);
        }
    }

    // ---- image tokens: leaky(images,0.01) @ W_img + b_img  (784 x 2048 x 512)
    k_wmma_gemm<<<dim3(512/BN, CDIV(784,BM)),256,0,stream>>>(
        images, nullptr, 2048, W_img, 512, b_img, IMG, 512, 784, 512, 2048, 1);

    // ---- final token assembly (16, 177, 512)
    k_scatter_out<<<CDIV(16*177*512,256),256,0,stream>>>(IMG, X1, batched, out);
}
